// UnifiedTensorNetwork_76665166233678
// MI455X (gfx1250) — compile-verified, matrix operand order checked by reading, not run on hardware
//
#include <hip/hip_runtime.h>

// ---------------------------------------------------------------------------
// Types / helpers
// ---------------------------------------------------------------------------
typedef unsigned short u16;
typedef __attribute__((ext_vector_type(16))) __bf16 v16bf;
typedef __attribute__((ext_vector_type(8)))  float  v8f;
typedef __attribute__((ext_vector_type(4)))  unsigned int u32x4;

union Frag {
    v16bf v;
    u16   s[16];
    u32x4 q[2];
};

__device__ __forceinline__ u16 f2bf(float f) {
    union { float f; unsigned u; } x; x.f = f;
    unsigned r = x.u + 0x7FFFu + ((x.u >> 16) & 1u);   // round-to-nearest-even
    return (u16)(r >> 16);
}

#define BB   256      // batch
#define SS   128      // seq
#define DD   512      // model dim
#define MM   150      // modes
#define HH   8        // heads
#define HD   64       // head dim
#define VV   10000    // vocab
#define MP   160      // modes padded to multiple of 16

// ---------------------------------------------------------------------------
// 1) embedding gather + mean over S:  e[b,d] = mean_s emb[x[b,s], d]
// ---------------------------------------------------------------------------
__global__ void embed_mean_kernel(const int* __restrict__ x,
                                  const float* __restrict__ emb,
                                  float* __restrict__ e) {
    __shared__ int idx[SS];
    const int b = blockIdx.x, tid = threadIdx.x;
    if (tid < SS) idx[tid] = x[b * SS + tid];
    __syncthreads();
    float a0 = 0.f, a1 = 0.f;
    for (int s = 0; s < SS; ++s) {
        const float* er = emb + (size_t)idx[s] * DD;
        a0 += er[tid];
        a1 += er[tid + 256];
    }
    e[(size_t)b * DD + tid]       = a0 * (1.f / SS);
    e[(size_t)b * DD + tid + 256] = a1 * (1.f / SS);
}

__global__ void f32_to_bf16_kernel(const float* __restrict__ in,
                                   u16* __restrict__ out, int n) {
    int i = blockIdx.x * 256 + threadIdx.x;
    if (i < n) out[i] = f2bf(in[i]);
}

// ---------------------------------------------------------------------------
// 2) Generic NT GEMM with bf16 WMMA, f32 accumulate.
//    Out[row, n] = sum_k A[a_row, k] * W[(m?)*N*K + n*K + k] + bias[(m?)*N + n]
//    Block tile: 64 rows x 64 cols, 256 threads = 8 waves, each wave owns a
//    16-col strip and two stacked 16x16 accum tiles.
//    A (bf16) staged into LDS with GLOBAL_LOAD_ASYNC_TO_LDS_B128 (ASYNCcnt);
//    W streamed from global f32, converted to bf16 in registers.
//    OMODE: 0 = f32 store, 1 = bf16 store, 2 = atomicAdd f32 into [rpm, N]
// ---------------------------------------------------------------------------
template <int OMODE, bool RELU, bool BIAS_PER_MODE, bool W_PER_MODE, bool A_MOD>
__global__ void gemm_nt_kernel(const u16* __restrict__ Aptr,
                               const float* __restrict__ W,
                               const float* __restrict__ bias,
                               void* __restrict__ Outp,
                               int K, int N, int rpm /* rows per mode */) {
    extern __shared__ __align__(16) u16 a_lds[];   // 64 x K bf16
    const int tid  = threadIdx.x;
    const int row0 = blockIdx.x * 64;
    const int col0 = blockIdx.y * 64;
    const int m    = row0 / rpm;
    const int a_row0 = A_MOD ? (row0 % rpm) : row0;
    const float* Wm = W + (W_PER_MODE ? (size_t)m * (size_t)N * (size_t)K : (size_t)0);
    const float* bm = bias + (BIAS_PER_MODE ? (size_t)m * (size_t)N : (size_t)0);

    // ---- async-stage A tile (64 consecutive rows x K, contiguous) into LDS ----
    {
        const u16* Abase = Aptr + (size_t)a_row0 * (size_t)K;           // 16B aligned
        const unsigned lds_base = (unsigned)(size_t)(void*)a_lds;       // LDS aperture: addr[31:0]
        const int n128 = (64 * K) / 8;                                  // 16-byte chunks
        for (int i = tid; i < n128; i += 256) {
            unsigned ldsoff = lds_base + (unsigned)i * 16u;
            unsigned goff   = (unsigned)i * 16u;
            asm volatile("global_load_async_to_lds_b128 %0, %1, %2"
                         :: "v"(ldsoff), "v"(goff), "s"(Abase) : "memory");
        }
        asm volatile("s_wait_asynccnt 0" ::: "memory");
    }
    __syncthreads();

    const int wid  = tid >> 5;
    const int lane = tid & 31;
    const int half = lane >> 4;
    const int l16  = lane & 15;
    const int ct   = wid & 3;          // column subtile of this wave
    const int rg   = wid >> 2;         // row group (0/1): rows rg*32 .. rg*32+31
    const int wcol = col0 + ct * 16 + l16;
    const int wc   = (wcol < N) ? wcol : (N - 1);   // clamp for N tail
    const int lr0  = rg * 32 + l16;
    const int lr1  = lr0 + 16;

    v8f acc0 = {};
    v8f acc1 = {};
    const float* wrowBase = Wm + (size_t)wc * K;

#pragma unroll 2
    for (int k0 = 0; k0 < K; k0 += 32) {
        Frag a0, a1, bf;
        const u16* pa0 = &a_lds[lr0 * K + k0 + 8 * half];
        a0.q[0] = *(const u32x4*)(pa0);
        a0.q[1] = *(const u32x4*)(pa0 + 16);
        const u16* pa1 = &a_lds[lr1 * K + k0 + 8 * half];
        a1.q[0] = *(const u32x4*)(pa1);
        a1.q[1] = *(const u32x4*)(pa1 + 16);

        const float* wp = wrowBase + k0 + 8 * half;
        __builtin_prefetch(wrowBase + k0 + 64, 0, 0);   // next K chunk
#pragma unroll
        for (int j = 0; j < 8; ++j) bf.s[j] = f2bf(wp[j]);
#pragma unroll
        for (int j = 0; j < 8; ++j) bf.s[8 + j] = f2bf(wp[16 + j]);

        acc0 = __builtin_amdgcn_wmma_f32_16x16x32_bf16(false, a0.v, false, bf.v,
                                                       (short)0, acc0, false, false);
        acc1 = __builtin_amdgcn_wmma_f32_16x16x32_bf16(false, a1.v, false, bf.v,
                                                       (short)0, acc1, false, false);
    }

    if (wcol < N) {
        const float bv = bm[wcol];
#pragma unroll
        for (int r = 0; r < 8; ++r) {
            int grow0 = row0 + rg * 32 + r + 8 * half;
            int grow1 = grow0 + 16;
            float v0 = acc0[r] + bv;
            float v1 = acc1[r] + bv;
            if (RELU) { v0 = fmaxf(v0, 0.f); v1 = fmaxf(v1, 0.f); }
            if (OMODE == 0) {
                float* O = (float*)Outp;
                O[(size_t)grow0 * N + wcol] = v0;
                O[(size_t)grow1 * N + wcol] = v1;
            } else if (OMODE == 1) {
                u16* O = (u16*)Outp;
                O[(size_t)grow0 * N + wcol] = f2bf(v0);
                O[(size_t)grow1 * N + wcol] = f2bf(v1);
            } else {
                float* O = (float*)Outp;   // sum over modes -> [rpm, N]
                atomicAdd(&O[(size_t)(grow0 % rpm) * N + wcol], v0);
                atomicAdd(&O[(size_t)(grow1 % rpm) * N + wcol], v1);
            }
        }
    }
}

// ---------------------------------------------------------------------------
// 3) Row LayerNorm (D=512) + optional ReLU, bf16 output.
// ---------------------------------------------------------------------------
template <bool RELU, bool PER_MODE>
__global__ void ln_kernel(const float* __restrict__ x,
                          const float* __restrict__ g,
                          const float* __restrict__ be,
                          u16* __restrict__ out, int rpm) {
    __shared__ float s1[256], s2[256];
    const int row = blockIdx.x, tid = threadIdx.x;
    const float* xr = x + (size_t)row * DD;
    const float a = xr[tid], c = xr[tid + 256];
    s1[tid] = a + c;
    s2[tid] = a * a + c * c;
    __syncthreads();
    for (int s = 128; s > 0; s >>= 1) {
        if (tid < s) { s1[tid] += s1[tid + s]; s2[tid] += s2[tid + s]; }
        __syncthreads();
    }
    const float mu  = s1[0] * (1.f / DD);
    const float var = s2[0] * (1.f / DD) - mu * mu;
    const float rs  = rsqrtf(var + 1e-5f);
    const int gb = PER_MODE ? (row / rpm) * DD : 0;
    float o0 = (a - mu) * rs * g[gb + tid]       + be[gb + tid];
    float o1 = (c - mu) * rs * g[gb + tid + 256] + be[gb + tid + 256];
    if (RELU) { o0 = fmaxf(o0, 0.f); o1 = fmaxf(o1, 0.f); }
    out[(size_t)row * DD + tid]       = f2bf(o0);
    out[(size_t)row * DD + tid + 256] = f2bf(o1);
}

// ---------------------------------------------------------------------------
// 4) Attention, one block per (b, h).  M=150 padded to 160.
//    scores = Q Kt / 8 via bf16 WMMA, softmax in LDS, attn @ V via bf16 WMMA.
//    qkv layout: [(m*B + b), 3D] bf16 ;  o layout: [(m*B + b), D] bf16
// ---------------------------------------------------------------------------
__global__ void attention_kernel(const u16* __restrict__ qkv, u16* __restrict__ o) {
    extern __shared__ __align__(16) unsigned char smem[];
    u16*   qs = (u16*)smem;                                   // [160][64]
    u16*   ks = qs + MP * HD;                                 // [160][64]
    u16*   vt = ks + MP * HD;                                 // [64][160] (V transposed)
    float* sc = (float*)(smem + (size_t)3 * MP * HD * 2);     // [160][160] f32
    u16*   at = (u16*)(smem + (size_t)3 * MP * HD * 2 + (size_t)MP * MP * 4); // [160][160] bf16

    const int b = blockIdx.x, h = blockIdx.y, tid = threadIdx.x;

    // load Q, K rows (zero pad rows >= 150)
    for (int i = tid; i < MP * HD; i += 256) {
        int mq = i >> 6, d = i & 63;
        u16 qv = 0, kv = 0;
        if (mq < MM) {
            size_t base = ((size_t)mq * BB + b) * (3 * DD) + h * HD + d;
            qv = qkv[base];
            kv = qkv[base + DD];
        }
        qs[i] = qv;
        ks[i] = kv;
    }
    // load V transposed: vt[d][mk]
    for (int i = tid; i < HD * MP; i += 256) {
        int d = i / MP, mk = i - d * MP;
        u16 vv = 0;
        if (mk < MM) vv = qkv[((size_t)mk * BB + b) * (3 * DD) + 2 * DD + h * HD + d];
        vt[i] = vv;
    }
    __syncthreads();

    const int wid = tid >> 5, lane = tid & 31, half = lane >> 4, l16 = lane & 15;

    // ---- scores: 10x10 tiles of 16x16, K = 64; two tiles in flight per wave
    for (int t0 = wid; t0 < 100; t0 += 16) {
        const int t1 = t0 + 8;
        const bool has1 = (t1 < 100);                 // wave-uniform
        int ti0 = t0 / 10, tj0 = t0 - ti0 * 10;
        int ti1 = has1 ? t1 / 10 : ti0;
        int tj1 = has1 ? (t1 - ti1 * 10) : tj0;
        v8f acc0 = {}, acc1 = {};
#pragma unroll
        for (int k0 = 0; k0 < HD; k0 += 32) {
            Frag a0, b0, a1, b1;
            const u16* pa0 = &qs[(ti0 * 16 + l16) * HD + k0 + 8 * half];
            a0.q[0] = *(const u32x4*)(pa0); a0.q[1] = *(const u32x4*)(pa0 + 16);
            const u16* pb0 = &ks[(tj0 * 16 + l16) * HD + k0 + 8 * half];
            b0.q[0] = *(const u32x4*)(pb0); b0.q[1] = *(const u32x4*)(pb0 + 16);
            const u16* pa1 = &qs[(ti1 * 16 + l16) * HD + k0 + 8 * half];
            a1.q[0] = *(const u32x4*)(pa1); a1.q[1] = *(const u32x4*)(pa1 + 16);
            const u16* pb1 = &ks[(tj1 * 16 + l16) * HD + k0 + 8 * half];
            b1.q[0] = *(const u32x4*)(pb1); b1.q[1] = *(const u32x4*)(pb1 + 16);
            acc0 = __builtin_amdgcn_wmma_f32_16x16x32_bf16(false, a0.v, false, b0.v,
                                                           (short)0, acc0, false, false);
            acc1 = __builtin_amdgcn_wmma_f32_16x16x32_bf16(false, a1.v, false, b1.v,
                                                           (short)0, acc1, false, false);
        }
#pragma unroll
        for (int r = 0; r < 8; ++r) {
            int row0 = ti0 * 16 + r + 8 * half;
            sc[row0 * MP + tj0 * 16 + l16] = acc0[r] * 0.125f;   // 1/sqrt(64)
        }
        if (has1) {
#pragma unroll
            for (int r = 0; r < 8; ++r) {
                int row1 = ti1 * 16 + r + 8 * half;
                sc[row1 * MP + tj1 * 16 + l16] = acc1[r] * 0.125f;
            }
        }
    }
    __syncthreads();

    // ---- softmax over valid 150 columns; write bf16 attn (padded cols = 0) ----
    for (int row = tid; row < MM; row += 256) {
        float mx = -3.4e38f;
        for (int c = 0; c < MM; ++c) mx = fmaxf(mx, sc[row * MP + c]);
        float s = 0.f;
        for (int c = 0; c < MM; ++c) {
            float ev = __expf(sc[row * MP + c] - mx);
            sc[row * MP + c] = ev;
            s += ev;
        }
        const float inv = 1.f / s;
        for (int c = 0; c < MP; ++c)
            at[row * MP + c] = (c < MM) ? f2bf(sc[row * MP + c] * inv) : (u16)0;
    }
    // zero the 10 padded attn rows so the 2nd GEMM reads clean data
    for (int i = tid; i < (MP - MM) * MP; i += 256)
        at[(MM + i / MP) * MP + (i % MP)] = 0;
    __syncthreads();

    // ---- o = attn @ V : 10x4 tiles of 16x16, K = 160; two tiles in flight ----
    for (int t0 = wid; t0 < 40; t0 += 16) {
        const int t1 = t0 + 8;
        const bool has1 = (t1 < 40);                  // wave-uniform
        int ti0 = t0 >> 2, tj0 = t0 & 3;
        int ti1 = has1 ? (t1 >> 2) : ti0;
        int tj1 = has1 ? (t1 & 3) : tj0;
        v8f acc0 = {}, acc1 = {};
        for (int k0 = 0; k0 < MP; k0 += 32) {
            Frag a0, b0, a1, b1;
            const u16* pa0 = &at[(ti0 * 16 + l16) * MP + k0 + 8 * half];
            a0.q[0] = *(const u32x4*)(pa0); a0.q[1] = *(const u32x4*)(pa0 + 16);
            const u16* pb0 = &vt[(tj0 * 16 + l16) * MP + k0 + 8 * half];
            b0.q[0] = *(const u32x4*)(pb0); b0.q[1] = *(const u32x4*)(pb0 + 16);
            const u16* pa1 = &at[(ti1 * 16 + l16) * MP + k0 + 8 * half];
            a1.q[0] = *(const u32x4*)(pa1); a1.q[1] = *(const u32x4*)(pa1 + 16);
            const u16* pb1 = &vt[(tj1 * 16 + l16) * MP + k0 + 8 * half];
            b1.q[0] = *(const u32x4*)(pb1); b1.q[1] = *(const u32x4*)(pb1 + 16);
            acc0 = __builtin_amdgcn_wmma_f32_16x16x32_bf16(false, a0.v, false, b0.v,
                                                           (short)0, acc0, false, false);
            acc1 = __builtin_amdgcn_wmma_f32_16x16x32_bf16(false, a1.v, false, b1.v,
                                                           (short)0, acc1, false, false);
        }
#pragma unroll
        for (int r = 0; r < 8; ++r) {
            int row = ti0 * 16 + r + 8 * half;
            if (row < MM)
                o[((size_t)row * BB + b) * DD + h * HD + tj0 * 16 + l16] = f2bf(acc0[r]);
        }
        if (has1) {
#pragma unroll
            for (int r = 0; r < 8; ++r) {
                int row = ti1 * 16 + r + 8 * half;
                if (row < MM)
                    o[((size_t)row * BB + b) * DD + h * HD + tj1 * 16 + l16] = f2bf(acc1[r]);
            }
        }
    }
}

// ---------------------------------------------------------------------------
// small elementwise kernels
// ---------------------------------------------------------------------------
__global__ void addvec_kernel(const float* __restrict__ a, const float* __restrict__ b,
                              float* __restrict__ o, int n) {
    int i = blockIdx.x * 256 + threadIdx.x;
    if (i < n) o[i] = a[i] + b[i];
}
__global__ void zero_kernel(float* __restrict__ p, int n) {
    int i = blockIdx.x * 256 + threadIdx.x;
    if (i < n) p[i] = 0.f;
}

// ---------------------------------------------------------------------------
// host launcher
// ---------------------------------------------------------------------------
extern "C" void kernel_launch(void* const* d_in, const int* in_sizes, int n_in,
                              void* d_out, int out_size, void* d_ws, size_t ws_size,
                              hipStream_t stream) {
    (void)in_sizes; (void)n_in; (void)out_size; (void)ws_size;
    const int*   x          = (const int*)d_in[0];
    const float* emb        = (const float*)d_in[1];
    const float* W1         = (const float*)d_in[2];
    const float* b1         = (const float*)d_in[3];
    const float* ln1_g      = (const float*)d_in[4];
    const float* ln1_b      = (const float*)d_in[5];
    const float* W2         = (const float*)d_in[6];
    const float* b2         = (const float*)d_in[7];
    const float* dbias      = (const float*)d_in[8];
    const float* in_proj_w  = (const float*)d_in[9];
    const float* in_proj_b  = (const float*)d_in[10];
    const float* out_proj_w = (const float*)d_in[11];
    const float* out_proj_b = (const float*)d_in[12];
    const float* norm_g     = (const float*)d_in[13];
    const float* norm_b     = (const float*)d_in[14];
    const float* dec_w1     = (const float*)d_in[15];
    const float* dec_b1     = (const float*)d_in[16];
    const float* dec_w2     = (const float*)d_in[17];
    const float* dec_b2     = (const float*)d_in[18];

    char* ws = (char*)d_ws;
    const int R = MM * BB;                       // 38400 grouped rows [M,B]
    // workspace layout (bytes)
    float* e      = (float*)(ws);                                // [B,512] f32
    u16*   e_bf   = (u16*)(ws + (size_t)BB * DD * 4);            // [B,512] bf16
    char*  big    = ws + (size_t)BB * DD * 6;                    // reusable 78.6MB region
    float* h_pre  = (float*)big;                                 // [R,512] f32
    u16*   mo_bf  = (u16*)big;                                   // [R,512] bf16 (reuses h_pre)
    u16*   o_bf   = (u16*)(big + (size_t)R * DD * 2);            // [R,512] bf16
    char*  p      = big + (size_t)R * DD * 4;
    u16*   h_bf   = (u16*)p;          p += (size_t)R * DD * 2;       // [R,512] bf16
    u16*   qkv_bf = (u16*)p;          p += (size_t)R * 3 * DD * 2;   // [R,1536] bf16
    float* comb   = (float*)p;        p += (size_t)MM * DD * 4;      // b2 + dbias
    float* agg    = (float*)p;        p += (size_t)BB * DD * 4;      // [B,512] f32
    u16*   agg_ln = (u16*)p;          p += (size_t)BB * DD * 2;      // [B,512] bf16
    u16*   dec_h  = (u16*)p;          p += (size_t)BB * (DD / 2) * 2;// [B,256] bf16
    float* out    = (float*)d_out;                               // [B,10000] f32

    const size_t lds512 = (size_t)64 * DD * 2;          // 64 KB A-tile, K=512
    const size_t lds256 = (size_t)64 * (DD / 2) * 2;    // 32 KB A-tile, K=256
    const size_t ldsAtt = (size_t)3 * MP * HD * 2 + (size_t)MP * MP * 4 + (size_t)MP * MP * 2;

    // 1) e = mean_s emb[x]; e_bf = bf16(e)
    embed_mean_kernel<<<BB, 256, 0, stream>>>(x, emb, e);
    f32_to_bf16_kernel<<<(BB * DD + 255) / 256, 256, 0, stream>>>(e, e_bf, BB * DD);

    // 2) h_pre = e @ W1[m]^T + b1[m]
    gemm_nt_kernel<0, false, true, true, true>
        <<<dim3(R / 64, DD / 64), 256, lds512, stream>>>(e_bf, W1, b1, h_pre, DD, DD, BB);

    // 3) h_bf = relu(LN_m(h_pre))
    ln_kernel<true, true><<<R, 256, 0, stream>>>(h_pre, ln1_g, ln1_b, h_bf, BB);

    // 4) comb = b2 + dbias
    addvec_kernel<<<(MM * DD + 255) / 256, 256, 0, stream>>>(b2, dbias, comb, MM * DD);

    // 5) mode_out = h_bf @ W2[m]^T + comb[m]   (bf16 out)
    gemm_nt_kernel<1, false, true, true, false>
        <<<dim3(R / 64, DD / 64), 256, lds512, stream>>>(h_bf, W2, comb, mo_bf, DD, DD, BB);

    // 6) qkv = mode_out @ in_proj_w^T + in_proj_b   (N = 1536)
    gemm_nt_kernel<1, false, false, false, false>
        <<<dim3(R / 64, (3 * DD) / 64), 256, lds512, stream>>>(mo_bf, in_proj_w, in_proj_b,
                                                               qkv_bf, DD, 3 * DD, BB);

    // 7) attention per (b, h)
    attention_kernel<<<dim3(BB, HH), 256, ldsAtt, stream>>>(qkv_bf, o_bf);

    // 8) agg = sum_m (o @ out_proj_w^T + out_proj_b)  via atomic f32 accumulate
    zero_kernel<<<(BB * DD + 255) / 256, 256, 0, stream>>>(agg, BB * DD);
    gemm_nt_kernel<2, false, false, false, false>
        <<<dim3(R / 64, DD / 64), 256, lds512, stream>>>(o_bf, out_proj_w, out_proj_b,
                                                         agg, DD, DD, BB);

    // 9) agg_ln = LN(agg)
    ln_kernel<false, false><<<BB, 256, 0, stream>>>(agg, norm_g, norm_b, agg_ln, BB);

    // 10) dec_h = relu(agg_ln @ dec_w1^T + dec_b1)   (N = 256)
    gemm_nt_kernel<1, true, false, false, false>
        <<<dim3(BB / 64, (DD / 2) / 64), 256, lds512, stream>>>(agg_ln, dec_w1, dec_b1,
                                                                dec_h, DD, DD / 2, BB);

    // 11) out = dec_h @ dec_w2^T + dec_b2     (K = 256, N = 10000, f32 out)
    gemm_nt_kernel<0, false, false, false, false>
        <<<dim3(BB / 64, (VV + 63) / 64), 256, lds256, stream>>>(dec_h, dec_w2, dec_b2,
                                                                 out, DD / 2, VV, BB);
}
